// CrossVit_10660108829094
// MI455X (gfx1250) — compile-verified, matrix operand order checked by reading, not run on hardware
//
#include <hip/hip_runtime.h>

// ---------------------------------------------------------------------------
// CrossViT fusion block for gfx1250 (MI455X): bf16 WMMA GEMMs, fp32 accumulate.
// ---------------------------------------------------------------------------

typedef __attribute__((ext_vector_type(16))) __bf16 bf16x16;
typedef __attribute__((ext_vector_type(8)))  float  f32x8;

static constexpr int N   = 196;   // tokens
static constexpr int E   = 256;   // embedding
static constexpr int H   = 8;     // heads
static constexpr int DH  = 32;    // head dim
static constexpr int HID = 1024;  // mlp hidden
static constexpr int T3  = 768;   // qkv width
static constexpr int NP  = 208;   // 13*16  (M padding for 16-row tiles)
static constexpr int KP  = 224;   // attention K padding (multiple of 32)
static constexpr int MT  = 13;    // number of 16-row tiles covering 196 rows

// ---- per-stream workspace layout (bytes; every block is 256B-aligned) ----
static constexpr size_t OFF_X16   = 0;                                   // NP x E   bf16
static constexpr size_t OFF_WQKVT = OFF_X16   + (size_t)NP * E * 2;      // T3 x E   bf16 (W^T)
static constexpr size_t OFF_WPT   = OFF_WQKVT + (size_t)T3 * E * 2;      // E  x E   bf16 (W^T)
static constexpr size_t OFF_W1T   = OFF_WPT   + (size_t)E  * E * 2;      // HID x E  bf16 (W^T)
static constexpr size_t OFF_W2T   = OFF_W1T   + (size_t)HID * E * 2;     // E x HID  bf16 (W^T)
static constexpr size_t OFF_NV    = OFF_W2T   + (size_t)E * HID * 2;     // N x E    f32 (LN1 out)
static constexpr size_t OFF_Q16   = OFF_NV    + (size_t)N * E * 4;       // H x NP x DH bf16
static constexpr size_t OFF_K16   = OFF_Q16   + (size_t)H * NP * DH * 2; // H x NP x DH bf16
static constexpr size_t OFF_VT16  = OFF_K16   + (size_t)H * NP * DH * 2; // H x DH x KP bf16
static constexpr size_t OFF_S     = OFF_VT16  + (size_t)H * DH * KP * 2; // H x NP x NP f32
static constexpr size_t OFF_ATT16 = OFF_S     + (size_t)H * NP * NP * 4; // H x NP x KP bf16
static constexpr size_t OFF_O16   = OFF_ATT16 + (size_t)H * NP * KP * 2; // NP x E   bf16
static constexpr size_t OFF_RV    = OFF_O16   + (size_t)NP * E * 2;      // N x E    f32
static constexpr size_t OFF_LV    = OFF_RV    + (size_t)N * E * 4;       // N x E    f32
static constexpr size_t OFF_LV16  = OFF_LV    + (size_t)N * E * 4;       // NP x E   bf16
static constexpr size_t OFF_H16   = OFF_LV16  + (size_t)NP * E * 2;      // NP x HID bf16
static constexpr size_t OFF_OV    = OFF_H16   + (size_t)NP * HID * 2;    // N x E    f32
static constexpr size_t STREAM_BYTES = OFF_OV + (size_t)N * E * 4;

#define WSP(T, s, off) ((T*)(ws + (size_t)(s) * STREAM_BYTES + (off)))

__device__ __forceinline__ unsigned short f2bf(float f) {
  unsigned u = __builtin_bit_cast(unsigned, f);
  u = u + 0x7fffu + ((u >> 16) & 1u);              // round-to-nearest-even
  return (unsigned short)(u >> 16);
}

// ---------------------------------------------------------------------------
// Wave-level 16x16 output tile, accumulating over K (multiple of 32).
// A : bf16, M-major rows, K contiguous (row stride lda)
// Bt: bf16, N-major rows, K contiguous (row stride ldb)  == B^T
// CDNA5 16-bit WMMA fragment layout: lane&15 selects the M (resp. N) row,
// lane>>4 selects K-halves {0-7,16-23} vs {8-15,24-31}; VGPR j holds K pair
// kk = (j>=4?16:0) + half*8 + (j&3)*2. Packed pairs load as single dwords.
// ---------------------------------------------------------------------------
__device__ __forceinline__ f32x8 wmma_kloop(const unsigned short* __restrict__ A, int lda,
                                            const unsigned short* __restrict__ Bt, int ldb,
                                            int K, f32x8 acc) {
  const int lane = threadIdx.x & 31;
  const int half = lane >> 4;
  const int r    = lane & 15;
  const unsigned short* ap = A  + (size_t)r * lda;
  const unsigned short* bp = Bt + (size_t)r * ldb;
  for (int k0 = 0; k0 < K; k0 += 32) {
    union { unsigned int u[8]; bf16x16 v; } fa, fb;
#pragma unroll
    for (int j = 0; j < 8; ++j) {
      const int kk = k0 + ((j >= 4) ? 16 : 0) + half * 8 + (j & 3) * 2;
      fa.u[j] = *(const unsigned int*)(ap + kk);
      fb.u[j] = *(const unsigned int*)(bp + kk);
    }
    acc = __builtin_amdgcn_wmma_f32_16x16x32_bf16(false, fa.v, false, fb.v,
                                                  (short)0, acc, false, false);
  }
  return acc;
}
// Result tile element j lives at row (tile_m*16 + j + 8*half), col (tile_n*16 + r).

// ---------------------------------------------------------------------------
// Utility kernels
// ---------------------------------------------------------------------------
__global__ void k_zero(uint4* p, size_t n16) {
  uint4 z = {0u, 0u, 0u, 0u};
  for (size_t i = (size_t)blockIdx.x * blockDim.x + threadIdx.x; i < n16;
       i += (size_t)gridDim.x * blockDim.x)
    p[i] = z;
}

// linear f32 -> bf16 (same row stride, padded rows stay zero)
__global__ void k_cvt(char* ws, const float* __restrict__ srcV,
                      const float* __restrict__ srcI, size_t off, int n) {
  const int s = blockIdx.z;
  const float* src = s ? srcI : srcV;
  unsigned short* dst = WSP(unsigned short, s, off);
  for (int i = blockIdx.x * blockDim.x + threadIdx.x; i < n; i += gridDim.x * blockDim.x)
    dst[i] = f2bf(src[i]);
}

// f32 (K x Nn, row-major) -> bf16 transposed (Nn x K, K contiguous)
__global__ void k_cvtT(char* ws, const float* __restrict__ srcV,
                       const float* __restrict__ srcI, size_t off, int K, int Nn) {
  const int s = blockIdx.z;
  const float* src = s ? srcI : srcV;
  unsigned short* dst = WSP(unsigned short, s, off);
  const int n = K * Nn;
  for (int i = blockIdx.x * blockDim.x + threadIdx.x; i < n; i += gridDim.x * blockDim.x) {
    const int k = i / Nn, c = i - k * Nn;
    dst[(size_t)c * K + k] = f2bf(src[i]);
  }
}

// LayerNorm over E=256 columns; optional fp32 and bf16 outputs.
__global__ void k_layernorm(const float* __restrict__ srcV, const float* __restrict__ srcI,
                            const float* __restrict__ wV, const float* __restrict__ bV,
                            const float* __restrict__ wI, const float* __restrict__ bI,
                            float* dFV, float* dFI,
                            unsigned short* dBV, unsigned short* dBI) {
  const int s = blockIdx.z;
  const float* src = s ? srcI : srcV;
  const float* w   = s ? wI : wV;
  const float* b   = s ? bI : bV;
  float* dF = s ? dFI : dFV;
  unsigned short* dB = s ? dBI : dBV;
  const int row = blockIdx.x, t = threadIdx.x;
  const float x = src[row * E + t];
  __shared__ float red[E];
  red[t] = x; __syncthreads();
  for (int o = 128; o > 0; o >>= 1) { if (t < o) red[t] += red[t + o]; __syncthreads(); }
  const float mean = red[0] * (1.0f / E);
  __syncthreads();
  const float d = x - mean;
  red[t] = d * d; __syncthreads();
  for (int o = 128; o > 0; o >>= 1) { if (t < o) red[t] += red[t + o]; __syncthreads(); }
  const float inv = rsqrtf(red[0] * (1.0f / E) + 1e-5f);
  const float y = d * inv * w[t] + b[t];
  if (dF) dF[row * E + t] = y;
  if (dB) dB[row * E + t] = f2bf(y);
}

// ---------------------------------------------------------------------------
// GEMM kernels (one wave = one 16x16 output tile)
// ---------------------------------------------------------------------------

// QKV: X16(196x256) @ WqkvT -> scatter into q16/k16 (H,NP,DH) and vT16 (H,DH,KP)
__global__ void k_qkv(char* ws, const float* __restrict__ bqV, const float* __restrict__ bqI) {
  const int mt = blockIdx.x, nt = blockIdx.y, s = blockIdx.z;
  const unsigned short* A  = WSP(unsigned short, s, OFF_X16)   + (size_t)mt * 16 * E;
  const unsigned short* Bt = WSP(unsigned short, s, OFF_WQKVT) + (size_t)nt * 16 * E;
  f32x8 acc = {0, 0, 0, 0, 0, 0, 0, 0};
  acc = wmma_kloop(A, E, Bt, E, E, acc);
  const float* bias = s ? bqI : bqV;
  const int lane = threadIdx.x & 31, half = lane >> 4, r = lane & 15;
  const int c   = nt * 16 + r;           // qkv column 0..767
  const int sid = c % 3;                 // 0=q 1=k 2=v
  const int d   = (c / 3) & (DH - 1);
  const int h   = c / (DH * 3);
  unsigned short* q16 = WSP(unsigned short, s, OFF_Q16);
  unsigned short* k16 = WSP(unsigned short, s, OFF_K16);
  unsigned short* vt  = WSP(unsigned short, s, OFF_VT16);
  const float bv = bias[c];
#pragma unroll
  for (int j = 0; j < 8; ++j) {
    const int m = mt * 16 + j + 8 * half;
    if (m < N) {
      const unsigned short hv = f2bf(acc[j] + bv);
      if (sid == 0)      q16[(size_t)h * NP * DH + (size_t)m * DH + d] = hv;
      else if (sid == 1) k16[(size_t)h * NP * DH + (size_t)m * DH + d] = hv;
      else               vt [(size_t)h * DH * KP + (size_t)d * KP + m] = hv;
    }
  }
}

// Scores (cross): S[s][h] = q16[s][h] @ k16[1-s][h]^T   (K = DH = 32 -> one WMMA)
__global__ void k_scores(char* ws) {
  const int mt = blockIdx.x, nt = blockIdx.y;
  const int s = blockIdx.z >> 3, h = blockIdx.z & 7;
  const unsigned short* A  = WSP(unsigned short, s,     OFF_Q16) + (size_t)h * NP * DH + (size_t)mt * 16 * DH;
  const unsigned short* Bt = WSP(unsigned short, s ^ 1, OFF_K16) + (size_t)h * NP * DH + (size_t)nt * 16 * DH;
  f32x8 acc = {0, 0, 0, 0, 0, 0, 0, 0};
  acc = wmma_kloop(A, DH, Bt, DH, DH, acc);
  float* S = WSP(float, s, OFF_S) + (size_t)h * NP * NP;
  const int lane = threadIdx.x & 31, half = lane >> 4, r = lane & 15;
#pragma unroll
  for (int j = 0; j < 8; ++j)
    S[(size_t)(mt * 16 + j + 8 * half) * NP + nt * 16 + r] = acc[j];
}

// Row softmax over 196 keys (with /16 scale) -> att16 bf16 (padded cols stay 0)
__global__ void k_softmax(char* ws) {
  const int row = blockIdx.x, h = blockIdx.y, s = blockIdx.z, t = threadIdx.x;
  const float* S = WSP(float, s, OFF_S) + (size_t)h * NP * NP + (size_t)row * NP;
  const float x = (t < N) ? S[t] * (1.0f / 16.0f) : -3.4e38f;
  __shared__ float red[256];
  red[t] = x; __syncthreads();
  for (int o = 128; o > 0; o >>= 1) { if (t < o) red[t] = fmaxf(red[t], red[t + o]); __syncthreads(); }
  const float mx = red[0];
  __syncthreads();
  const float e = (t < N) ? expf(x - mx) : 0.0f;
  red[t] = e; __syncthreads();
  for (int o = 128; o > 0; o >>= 1) { if (t < o) red[t] += red[t + o]; __syncthreads(); }
  const float inv = 1.0f / red[0];
  if (t < N)
    WSP(unsigned short, s, OFF_ATT16)[(size_t)h * NP * KP + (size_t)row * KP + t] = f2bf(e * inv);
}

// O = att16[s][h] @ v[1-s][h]  (K = KP = 224), packed into o16 (NP x 256)
__global__ void k_av(char* ws) {
  const int mt = blockIdx.x, dt = blockIdx.y;
  const int s = blockIdx.z >> 3, h = blockIdx.z & 7;
  const unsigned short* A  = WSP(unsigned short, s,     OFF_ATT16) + (size_t)h * NP * KP + (size_t)mt * 16 * KP;
  const unsigned short* Bt = WSP(unsigned short, s ^ 1, OFF_VT16)  + (size_t)h * DH * KP + (size_t)dt * 16 * KP;
  f32x8 acc = {0, 0, 0, 0, 0, 0, 0, 0};
  acc = wmma_kloop(A, KP, Bt, KP, KP, acc);
  unsigned short* o16 = WSP(unsigned short, s, OFF_O16);
  const int lane = threadIdx.x & 31, half = lane >> 4, r = lane & 15;
#pragma unroll
  for (int j = 0; j < 8; ++j) {
    const int m = mt * 16 + j + 8 * half;
    if (m < N) o16[(size_t)m * E + h * DH + dt * 16 + r] = f2bf(acc[j]);
  }
}

// Projection + residual: rv = nv + o16 @ WpT + bp
__global__ void k_proj(char* ws, const float* __restrict__ bpV, const float* __restrict__ bpI) {
  const int mt = blockIdx.x, nt = blockIdx.y, s = blockIdx.z;
  const unsigned short* A  = WSP(unsigned short, s, OFF_O16) + (size_t)mt * 16 * E;
  const unsigned short* Bt = WSP(unsigned short, s, OFF_WPT) + (size_t)nt * 16 * E;
  f32x8 acc = {0, 0, 0, 0, 0, 0, 0, 0};
  acc = wmma_kloop(A, E, Bt, E, E, acc);
  const float* bias = s ? bpI : bpV;
  const float* nv = WSP(float, s, OFF_NV);
  float* rv = WSP(float, s, OFF_RV);
  const int lane = threadIdx.x & 31, half = lane >> 4, r = lane & 15;
  const int c = nt * 16 + r;
  const float bv = bias[c];
#pragma unroll
  for (int j = 0; j < 8; ++j) {
    const int m = mt * 16 + j + 8 * half;
    if (m < N) rv[(size_t)m * E + c] = acc[j] + bv + nv[(size_t)m * E + c];
  }
}

// FF1: h16 = gelu(lv16 @ W1T + b1)  (exact erf gelu)
__global__ void k_ff1(char* ws, const float* __restrict__ b1V, const float* __restrict__ b1I) {
  const int mt = blockIdx.x, nt = blockIdx.y, s = blockIdx.z;
  const unsigned short* A  = WSP(unsigned short, s, OFF_LV16) + (size_t)mt * 16 * E;
  const unsigned short* Bt = WSP(unsigned short, s, OFF_W1T)  + (size_t)nt * 16 * E;
  f32x8 acc = {0, 0, 0, 0, 0, 0, 0, 0};
  acc = wmma_kloop(A, E, Bt, E, E, acc);
  const float* bias = s ? b1I : b1V;
  unsigned short* h16 = WSP(unsigned short, s, OFF_H16);
  const int lane = threadIdx.x & 31, half = lane >> 4, r = lane & 15;
  const int c = nt * 16 + r;
  const float bv = bias[c];
#pragma unroll
  for (int j = 0; j < 8; ++j) {
    const int m = mt * 16 + j + 8 * half;
    if (m < N) {
      const float x = acc[j] + bv;
      const float g = 0.5f * x * (1.0f + erff(x * 0.70710678118654752f));
      h16[(size_t)m * HID + c] = f2bf(g);
    }
  }
}

// FF2 + residual: ov = h16 @ W2T + b2 + lv   (K = 1024)
__global__ void k_ff2(char* ws, const float* __restrict__ b2V, const float* __restrict__ b2I) {
  const int mt = blockIdx.x, nt = blockIdx.y, s = blockIdx.z;
  const unsigned short* A  = WSP(unsigned short, s, OFF_H16) + (size_t)mt * 16 * HID;
  const unsigned short* Bt = WSP(unsigned short, s, OFF_W2T) + (size_t)nt * 16 * HID;
  f32x8 acc = {0, 0, 0, 0, 0, 0, 0, 0};
  acc = wmma_kloop(A, HID, Bt, HID, HID, acc);
  const float* bias = s ? b2I : b2V;
  const float* lv = WSP(float, s, OFF_LV);
  float* ov = WSP(float, s, OFF_OV);
  const int lane = threadIdx.x & 31, half = lane >> 4, r = lane & 15;
  const int c = nt * 16 + r;
  const float bv = bias[c];
#pragma unroll
  for (int j = 0; j < 8; ++j) {
    const int m = mt * 16 + j + 8 * half;
    if (m < N) ov[(size_t)m * E + c] = acc[j] + bv + lv[(size_t)m * E + c];
  }
}

// Pixel-shuffle recon: out[ch, p] with ch0 <- ir stream, ch1 <- vis stream.
// p = a*3136 + b*196 + (i*14+j); value = src[(i*14+j)*256 + a*16 + b]
__global__ void k_recon(char* ws, float* __restrict__ out) {
  const float* ovV = WSP(float, 0, OFF_OV);
  const float* ovI = WSP(float, 1, OFF_OV);
  const int total = 2 * 224 * 224;
  for (int idx = blockIdx.x * blockDim.x + threadIdx.x; idx < total;
       idx += gridDim.x * blockDim.x) {
    const int ch = idx / (224 * 224);
    const int p  = idx - ch * (224 * 224);
    const int a  = p / 3136;
    const int rem = p - a * 3136;
    const int b  = rem / 196;
    const int n  = rem - b * 196;
    const float* src = (ch == 0) ? ovI : ovV;
    out[idx] = src[(size_t)n * E + a * 16 + b];
  }
}

// ---------------------------------------------------------------------------
extern "C" void kernel_launch(void* const* d_in, const int* in_sizes, int n_in,
                              void* d_out, int out_size, void* d_ws, size_t ws_size,
                              hipStream_t stream) {
  char* ws = (char*)d_ws;
  const float* vis    = (const float*)d_in[0];
  const float* ir     = (const float*)d_in[1];
  const float* ln1v_w = (const float*)d_in[2];
  const float* ln1v_b = (const float*)d_in[3];
  const float* ln1i_w = (const float*)d_in[4];
  const float* ln1i_b = (const float*)d_in[5];
  const float* ln2v_w = (const float*)d_in[6];
  const float* ln2v_b = (const float*)d_in[7];
  const float* ln2i_w = (const float*)d_in[8];
  const float* ln2i_b = (const float*)d_in[9];
  const float* Wqkv_v = (const float*)d_in[10];
  const float* bqkv_v = (const float*)d_in[11];
  const float* Wqkv_i = (const float*)d_in[12];
  const float* bqkv_i = (const float*)d_in[13];
  const float* Wp_v   = (const float*)d_in[14];
  const float* bp_v   = (const float*)d_in[15];
  const float* Wp_i   = (const float*)d_in[16];
  const float* bp_i   = (const float*)d_in[17];
  const float* W1v    = (const float*)d_in[18];
  const float* b1v    = (const float*)d_in[19];
  const float* W2v    = (const float*)d_in[20];
  const float* b2v    = (const float*)d_in[21];
  const float* W1i    = (const float*)d_in[22];
  const float* b1i    = (const float*)d_in[23];
  const float* W2i    = (const float*)d_in[24];
  const float* b2i    = (const float*)d_in[25];

  // 0) zero the workspace (tile-padding regions must read as 0 for WMMA)
  const size_t n16 = (2 * STREAM_BYTES) / 16;
  k_zero<<<2048, 256, 0, stream>>>((uint4*)ws, n16);

  // 1) bf16 conversions: activations row-major, weights transposed (N-major)
  k_cvt <<<dim3(256, 1, 2), 256, 0, stream>>>(ws, vis, ir, OFF_X16, N * E);
  k_cvtT<<<dim3(512, 1, 2), 256, 0, stream>>>(ws, Wqkv_v, Wqkv_i, OFF_WQKVT, E, T3);
  k_cvtT<<<dim3(256, 1, 2), 256, 0, stream>>>(ws, Wp_v,   Wp_i,   OFF_WPT,   E, E);
  k_cvtT<<<dim3(512, 1, 2), 256, 0, stream>>>(ws, W1v,    W1i,    OFF_W1T,   E, HID);
  k_cvtT<<<dim3(512, 1, 2), 256, 0, stream>>>(ws, W2v,    W2i,    OFF_W2T,   HID, E);

  // 2) LN1 (residual path only, fp32)
  k_layernorm<<<dim3(N, 1, 2), 256, 0, stream>>>(
      vis, ir, ln1v_w, ln1v_b, ln1i_w, ln1i_b,
      (float*)(ws + OFF_NV), (float*)(ws + STREAM_BYTES + OFF_NV),
      (unsigned short*)nullptr, (unsigned short*)nullptr);

  // 3) QKV GEMMs (from RAW embeddings, per reference)
  k_qkv<<<dim3(MT, T3 / 16, 2), 32, 0, stream>>>(ws, bqkv_v, bqkv_i);

  // 4) cross-attention scores: q[s] x k[1-s]
  k_scores<<<dim3(MT, MT, 16), 32, 0, stream>>>(ws);

  // 5) softmax (/16 scale folded in)
  k_softmax<<<dim3(N, H, 2), 256, 0, stream>>>(ws);

  // 6) att[s] x v[1-s]
  k_av<<<dim3(MT, DH / 16, 16), 32, 0, stream>>>(ws);

  // 7) projection + LN1 residual
  k_proj<<<dim3(MT, E / 16, 2), 32, 0, stream>>>(ws, bp_v, bp_i);

  // 8) LN2 -> lv (fp32) + lv16 (bf16)
  k_layernorm<<<dim3(N, 1, 2), 256, 0, stream>>>(
      (const float*)(ws + OFF_RV), (const float*)(ws + STREAM_BYTES + OFF_RV),
      ln2v_w, ln2v_b, ln2i_w, ln2i_b,
      (float*)(ws + OFF_LV), (float*)(ws + STREAM_BYTES + OFF_LV),
      (unsigned short*)(ws + OFF_LV16), (unsigned short*)(ws + STREAM_BYTES + OFF_LV16));

  // 9) MLP
  k_ff1<<<dim3(MT, HID / 16, 2), 32, 0, stream>>>(ws, b1v, b1i);
  k_ff2<<<dim3(MT, E / 16, 2), 32, 0, stream>>>(ws, b2v, b2i);

  // 10) pixel-shuffle recon + channel concat [ir, vis]
  k_recon<<<256, 256, 0, stream>>>(ws, (float*)d_out);
}
